// SAGE_14173392077064
// MI455X (gfx1250) — matrix operand order repaired and checked by reading, not compiled, not run on hardware
//
#include <hip/hip_runtime.h>

#define D 128
#define EPSV 1e-5f

typedef _Float16 v16h __attribute__((ext_vector_type(16)));
typedef _Float16 v2h  __attribute__((ext_vector_type(2)));
typedef float    v8f  __attribute__((ext_vector_type(8)));

union HF16 { v16h v; _Float16 h[16]; };
union F8   { v8f  v; float f[8]; };

__global__ void zero_f32(float* p, long n) {
  long i = (long)blockIdx.x * blockDim.x + threadIdx.x;
  if (i < n) p[i] = 0.f;
}

__global__ void deg_kernel(const int* __restrict__ dst, float* __restrict__ deg, int E) {
  int e = blockIdx.x * blockDim.x + threadIdx.x;
  if (e < E) atomicAdd(&deg[dst[e]], 1.0f);
}

__global__ void invdeg_kernel(const float* __restrict__ deg, float* __restrict__ invdeg, int n) {
  int i = blockIdx.x * blockDim.x + threadIdx.x;
  if (i < n) invdeg[i] = 1.0f / fmaxf(deg[i], 1.0f);
}

// One wave (32 lanes) per edge: lane loads float4 of h[src] row, scatters with f32 atomics.
// Both h and agg are L2-resident (25.6 MB each << 192 MB L2).
__global__ void agg_kernel(const float* __restrict__ h, const int* __restrict__ src,
                           const int* __restrict__ dst, float* __restrict__ agg, int E) {
  int gid  = blockIdx.x * blockDim.x + threadIdx.x;
  int e    = gid >> 5;
  int lane = gid & 31;
  if (e >= E) return;
  int s = src[e], d = dst[e];
  const float4 v = *(const float4*)&h[(size_t)s * D + lane * 4];
  float* a = &agg[(size_t)d * D + lane * 4];
  atomicAdd(a + 0, v.x);
  atomicAdd(a + 1, v.y);
  atomicAdd(a + 2, v.z);
  atomicAdd(a + 3, v.w);
}

// Fused dual-GEMM: out = h@Wself + (agg*invdeg)@Wneigh + bias + h  (+ BN stats)
// fp32 accuracy via 3-term f16 split on V_WMMA_F32_16X16X32_F16.
// Block = 8 wave32 = one 16-row strip; wave w owns column tile [16w,16w+16).
// W (hi/lo, transposed) AND the A strip (hi/lo) are staged in LDS once per block,
// so the inner loop is DS loads + WMMA only.
__global__ __launch_bounds__(256) void sage_gemm(
    const float* __restrict__ h, const float* __restrict__ agg,
    const float* __restrict__ Wself, const float* __restrict__ Wneigh,
    const float* __restrict__ bias, const float* __restrict__ invdeg,
    float* __restrict__ out, float* __restrict__ colsum, float* __restrict__ colsq,
    int doStats)
{
  extern __shared__ _Float16 smem[];           // 144 KB total
  _Float16* WhS = smem;                        // [n][k] hi(Wself)   16384
  _Float16* WlS = smem + 16384;                // lo(Wself)
  _Float16* WhN = smem + 32768;                // hi(Wneigh)
  _Float16* WlN = smem + 49152;                // lo(Wneigh)
  _Float16* AhH = smem + 65536;                // [r][k] hi(h strip)  2048
  _Float16* AhL = smem + 67584;                // lo(h strip)
  _Float16* AnH = smem + 69632;                // hi(agg strip, scaled)
  _Float16* AnL = smem + 71680;                // lo(agg strip, scaled)

  int tid  = threadIdx.x;
  int row0 = blockIdx.x * 16;

  // Stage W row-major [k][n] -> LDS [n][k], split into hi/lo f16.
  for (int i = tid; i < D * D; i += 256) {
    int k = i >> 7, n = i & (D - 1);
    float ws = Wself[i];
    _Float16 wsh = (_Float16)ws;
    WhS[n * D + k] = wsh;
    WlS[n * D + k] = (_Float16)(ws - (float)wsh);
    float wn = Wneigh[i];
    _Float16 wnh = (_Float16)wn;
    WhN[n * D + k] = wnh;
    WlN[n * D + k] = (_Float16)(wn - (float)wnh);
  }
  // Stage the 16x128 A strips (shared by all 8 waves): split once, reuse 8x.
  for (int i = tid; i < 16 * D; i += 256) {
    int r = i >> 7;
    float xv = h[(size_t)(row0 + r) * D + (i & (D - 1))];
    _Float16 xh = (_Float16)xv;
    AhH[i] = xh;
    AhL[i] = (_Float16)(xv - (float)xh);
    float av = agg[(size_t)(row0 + r) * D + (i & (D - 1))] * invdeg[row0 + r];
    _Float16 ah = (_Float16)av;
    AnH[i] = ah;
    AnL[i] = (_Float16)(av - (float)ah);
  }
  __syncthreads();

  int wave   = tid >> 5;
  int lane   = tid & 31;
  int m      = lane & 15;
  int hiHalf = lane >> 4;
  int col0   = wave * 16;

  v8f acc = {};

#pragma unroll
  for (int k0 = 0; k0 < D; k0 += 32) {
    HF16 ash, asl, anh, anl, bsh, bsl, bnh, bnl;
    // A fragment: 16x32 f16; lane(0-15)=row m, K 0-7/16-23; lane(16-31)=row m, K 8-15/24-31
#pragma unroll
    for (int v = 0; v < 8; ++v) {
      int kk  = k0 + ((v & 4) ? 16 : 0) + hiHalf * 8 + (v & 3) * 2;
      int off = m * D + kk;                        // contiguous pair -> ds_load_b32
      *(v2h*)&ash.h[2*v] = *(const v2h*)&AhH[off];
      *(v2h*)&asl.h[2*v] = *(const v2h*)&AhL[off];
      *(v2h*)&anh.h[2*v] = *(const v2h*)&AnH[off];
      *(v2h*)&anl.h[2*v] = *(const v2h*)&AnL[off];
    }
    // B fragment: 32x16 f16; lane(0-15)=col n, K 0-7/16-23; lane(16-31)=col n, K 8-15/24-31
#pragma unroll
    for (int v = 0; v < 8; ++v) {
      int kk  = k0 + ((v & 4) ? 16 : 0) + hiHalf * 8 + (v & 3) * 2;
      int off = (col0 + m) * D + kk;
      *(v2h*)&bsh.h[2*v] = *(const v2h*)&WhS[off];
      *(v2h*)&bsl.h[2*v] = *(const v2h*)&WlS[off];
      *(v2h*)&bnh.h[2*v] = *(const v2h*)&WhN[off];
      *(v2h*)&bnl.h[2*v] = *(const v2h*)&WlN[off];
    }
    acc = __builtin_amdgcn_wmma_f32_16x16x32_f16(false, ash.v, false, bsh.v, (short)0, acc, false, false);
    acc = __builtin_amdgcn_wmma_f32_16x16x32_f16(false, asl.v, false, bsh.v, (short)0, acc, false, false);
    acc = __builtin_amdgcn_wmma_f32_16x16x32_f16(false, ash.v, false, bsl.v, (short)0, acc, false, false);
    acc = __builtin_amdgcn_wmma_f32_16x16x32_f16(false, anh.v, false, bnh.v, (short)0, acc, false, false);
    acc = __builtin_amdgcn_wmma_f32_16x16x32_f16(false, anl.v, false, bnh.v, (short)0, acc, false, false);
    acc = __builtin_amdgcn_wmma_f32_16x16x32_f16(false, anh.v, false, bnl.v, (short)0, acc, false, false);
  }

  // C/D layout: VGPR v -> row v (lanes 0-15) / row v+8 (lanes 16-31), col = lane&15
  F8 a; a.v = acc;
  int col  = col0 + m;
  float bc = bias[col];
  float s = 0.f, ss = 0.f;
#pragma unroll
  for (int v = 0; v < 8; ++v) {
    int row   = row0 + v + 8 * hiHalf;
    float val = a.f[v] + bc + h[(size_t)row * D + col];   // bias + residual
    out[(size_t)row * D + col] = val;
    s += val; ss += val * val;
  }
  if (doStats) {                                   // BN stats for free in epilogue
    atomicAdd(&colsum[col], s);
    atomicAdd(&colsq[col], ss);
  }
}

__global__ void bn_relu_kernel(float* __restrict__ hn, const float* __restrict__ colsum,
                               const float* __restrict__ colsq, const float* __restrict__ gamma,
                               const float* __restrict__ beta, long total, float invN) {
  long i = (long)blockIdx.x * blockDim.x + threadIdx.x;
  if (i >= total) return;
  int j = (int)(i & (D - 1));
  float mu  = colsum[j] * invN;
  float var = colsq[j] * invN - mu * mu;
  float x = (hn[i] - mu) * rsqrtf(var + EPSV) * gamma[j] + beta[j];
  hn[i] = fmaxf(x, 0.f);
}

extern "C" void kernel_launch(void* const* d_in, const int* in_sizes, int n_in,
                              void* d_out, int out_size, void* d_ws, size_t ws_size,
                              hipStream_t stream) {
  const float* feat   = (const float*)d_in[0];
  const int*   src    = (const int*)d_in[1];
  const int*   dst    = (const int*)d_in[2];
  const float* Wself  = (const float*)d_in[3];
  const float* Wneigh = (const float*)d_in[4];
  const float* bias   = (const float*)d_in[5];
  const float* gamma  = (const float*)d_in[6];
  const float* beta   = (const float*)d_in[7];
  const int N = in_sizes[0] / D;   // 50000 (divisible by 16)
  const int E = in_sizes[1];       // 800000

  float* ws = (float*)d_ws;
  size_t nd = (size_t)N * D;
  float* agg    = ws;              // [N,D]
  float* bufA   = agg + nd;        // [N,D]
  float* bufB   = bufA + nd;       // [N,D]
  float* deg    = bufB + nd;       // [N]
  float* invdeg = deg + N;         // [N]
  float* colsum = invdeg + N;      // [D]
  float* colsq  = colsum + D;      // [D] (contiguous with colsum)

  zero_f32<<<(N + 255) / 256, 256, 0, stream>>>(deg, N);
  deg_kernel<<<(E + 255) / 256, 256, 0, stream>>>(dst, deg, E);
  invdeg_kernel<<<(N + 255) / 256, 256, 0, stream>>>(deg, invdeg, N);

  size_t smemBytes = (4 * (size_t)D * D + 4 * 16 * D) * sizeof(_Float16); // 144 KB
  int aggBlocks = (int)((((size_t)E * 32) + 255) / 256);
  long total = (long)nd;

  const float* hin = feat;
  for (int layer = 0; layer < 3; ++layer) {
    float* hout = (layer == 0) ? bufA : (layer == 1) ? bufB : (float*)d_out;
    int doStats = (layer < 2);
    zero_f32<<<(int)((nd + 255) / 256), 256, 0, stream>>>(agg, (long)nd);
    agg_kernel<<<aggBlocks, 256, 0, stream>>>(hin, src, dst, agg, E);
    if (doStats) zero_f32<<<1, 256, 0, stream>>>(colsum, 2L * D);
    sage_gemm<<<N / 16, 256, smemBytes, stream>>>(
        hin, agg, Wself + (size_t)layer * D * D, Wneigh + (size_t)layer * D * D,
        bias + layer * D, invdeg, hout, colsum, colsq, doStats);
    if (doStats)
      bn_relu_kernel<<<(int)((total + 255) / 256), 256, 0, stream>>>(
          hout, colsum, colsq, gamma + layer * D, beta + layer * D, total, 1.0f / (float)N);
    hin = hout;
  }
}
